// Attention_38293928411140
// MI455X (gfx1250) — compile-verified, matrix operand order checked by reading, not run on hardware
//
#include <hip/hip_runtime.h>
#include <hip/hip_bf16.h>

// Problem constants (match reference setup_inputs)
#define S_LEN 4096
#define HID   512
#define NH    8
#define HD    64
#define NREG  32

typedef __attribute__((ext_vector_type(16))) _Float16 v16h;
typedef __attribute__((ext_vector_type(8)))  _Float16 v8h;
typedef __attribute__((ext_vector_type(4)))  _Float16 v4h;
typedef __attribute__((ext_vector_type(8)))  float    v8f;
typedef __attribute__((ext_vector_type(4)))  unsigned int u32x4;
typedef __attribute__((ext_vector_type(8)))  int      i32x8;
typedef __attribute__((ext_vector_type(4)))  int      i32x4;

__device__ __forceinline__ v8f wmma_f16(v16h a, v16h b, v8f c) {
  // D = A(16x32 f16) * B(32x16 f16) + C(16x16 f32)
  return __builtin_amdgcn_wmma_f32_16x16x32_f16(
      /*neg_a=*/false, a, /*neg_b=*/false, b,
      /*c_mod=*/(short)0, c, /*reuse_a=*/false, /*reuse_b=*/false);
}

// CDNA5 16-bit A/B fragment: per lane, two contiguous 8-half runs.
// lanes 0-15 : elems 0-7 -> K=k0+0..7,  elems 8-15 -> K=k0+16..23
// lanes 16-31: elems 0-7 -> K=k0+8..15, elems 8-15 -> K=k0+24..31
template <typename PT>
__device__ __forceinline__ v16h load_frag(PT p, int ld, int r0, int k0, int ln, int hi) {
  PT rp = p + (size_t)(r0 + ln) * (size_t)ld + k0;
  v8h a = *(const v8h*)(rp + (hi ? 8  : 0));
  v8h b = *(const v8h*)(rp + (hi ? 24 : 16));
  v16h r;
#pragma unroll
  for (int i = 0; i < 8; ++i) { r[i] = a[i]; r[i + 8] = b[i]; }
  return r;
}

// ---------------------------------------------------------------------------
// Tensor Data Mover: 2-D tile load (16-bit elements) Global -> LDS.
// Descriptor per CDNA5 ISA (D# groups 0/1; 2-D tensor,
// groups 2/3 zero). Tracked by TENSORcnt.
//   td0/st0 : tensor contiguous-dim length / row stride (elements)
//   td1     : tensor row count
//   tile0   : tile contiguous-dim length (elements), tile1: tile row count
// ---------------------------------------------------------------------------
__device__ __forceinline__ void tdm_load_2d(const void* gptr, unsigned lds_off,
                                            unsigned td0, unsigned td1,
                                            unsigned long long st0,
                                            unsigned tile0, unsigned tile1) {
  unsigned long long ga = (unsigned long long)(uintptr_t)gptr;
  u32x4 g0;
  g0[0] = 1u;                                                  // count=1, load
  g0[1] = lds_off;                                             // LDS byte addr
  g0[2] = (unsigned)ga;                                        // global addr lo
  g0[3] = ((unsigned)(ga >> 32) & 0x01FFFFFFu) | 0x80000000u;  // addr hi | type=2
  i32x8 g1;
  g1[0] = (int)(1u << 16);                                     // data_size = 2B
  g1[1] = (int)((td0 & 0xFFFFu) << 16);                        // tensor_dim0[15:0]
  g1[2] = (int)((td0 >> 16) | ((td1 & 0xFFFFu) << 16));        // td0 hi | td1 lo
  g1[3] = (int)((td1 >> 16) | ((tile0 & 0xFFFFu) << 16));      // td1 hi | tile0
  g1[4] = (int)(tile1 & 0xFFFFu);                              // tile1 (tile2=0)
  g1[5] = (int)(unsigned)st0;                                  // dim0 stride lo
  g1[6] = (int)(unsigned)(st0 >> 32);                          // stride hi (dim1 stride=0)
  g1[7] = 0;
  i32x4 z4 = {0, 0, 0, 0};
#if __clang_major__ >= 23
  i32x8 z8 = {0, 0, 0, 0, 0, 0, 0, 0};
  __builtin_amdgcn_tensor_load_to_lds(g0, g1, z4, z4, z8, 0);
#else
  __builtin_amdgcn_tensor_load_to_lds(g0, g1, z4, z4, 0);
#endif
}

__device__ __forceinline__ unsigned lds_offset_of(const void* p) {
  // Generic LDS pointers carry the LDS byte offset in the low 32 bits.
  return (unsigned)(uintptr_t)p;
}

// ---------------------------------------------------------------------------
// f32 -> f16 conversion (4 elements per thread)
// ---------------------------------------------------------------------------
__global__ void __launch_bounds__(256)
cvt_f32_to_f16(const float* __restrict__ in, _Float16* __restrict__ out, int n) {
  int i = (blockIdx.x * blockDim.x + threadIdx.x) * 4;
  if (i + 3 < n) {
    float4 v = *(const float4*)(in + i);
    v4h o;
    o[0] = (_Float16)v.x; o[1] = (_Float16)v.y;
    o[2] = (_Float16)v.z; o[3] = (_Float16)v.w;
    *(v4h*)(out + i) = o;
  }
}

// ---------------------------------------------------------------------------
// GEMM: C[s][o] = sum_h A[s][h] * W[o][h]   (A,W f16; acc f32)
// Workgroup = 4 waves; wave w owns a 16x64 tile (64 cols == one head).
// The 64x512 weight slab is TDM-staged into LDS once, shared by all waves.
// EPI: 0 f16 row-major | 1 f16 + fused RoPE | 2 f32 (final) | 3 f16 transposed
// ---------------------------------------------------------------------------
template <int EPI>
__global__ void __launch_bounds__(128)
gemm_kernel(const _Float16* __restrict__ A, const _Float16* __restrict__ W,
            void* __restrict__ Cout,
            const float* __restrict__ cosT, const float* __restrict__ sinT) {
  __shared__ _Float16 wt[64 * HID];   // 64 KB weight slab: rows = local n

  const int wave = threadIdx.x >> 5;
  const int lane = threadIdx.x & 31;
  const int ln = lane & 15, hi = lane >> 4;
  const int r0 = blockIdx.y * 64 + wave * 16;   // output rows (s)
  const int n0 = blockIdx.x * 64;               // output cols (o), head aligned

  if (wave == 0) {
    tdm_load_2d(W + (size_t)n0 * HID, lds_offset_of(&wt[0]),
                HID, HID, HID, /*tile0=*/HID, /*tile1=*/64);
    __builtin_amdgcn_s_wait_tensorcnt(0);
  }
  __syncthreads();

  v8f acc[4] = {v8f{}, v8f{}, v8f{}, v8f{}};

  for (int k0 = 0; k0 < HID; k0 += 32) {
    v16h af = load_frag(A, HID, r0, k0, ln, hi);
#pragma unroll
    for (int t = 0; t < 4; ++t) {
      v16h bf = load_frag((const _Float16*)wt, HID, 16 * t, k0, ln, hi);
      acc[t] = wmma_f16(af, bf, acc[t]);
    }
  }

  if constexpr (EPI == 1) {
    // Fused RoPE: tile t holds head-dim cols d = 16t+ln; partner d+-32 is t^2.
#pragma unroll
    for (int j = 0; j < 8; ++j) {
      int s = r0 + j + 8 * hi;
      if (s >= NREG) {
        int p = s - NREG;
        float nv[4];
#pragma unroll
        for (int t = 0; t < 4; ++t) {
          int d = 16 * t + ln;
          float c  = cosT[(size_t)p * HD + d];
          float sn = sinT[(size_t)p * HD + d];
          float v  = acc[t][j];
          float v2 = acc[t ^ 2][j];
          nv[t] = v * c + ((t < 2) ? -v2 : v2) * sn;
        }
#pragma unroll
        for (int t = 0; t < 4; ++t) acc[t][j] = nv[t];
      }
    }
  }

#pragma unroll
  for (int j = 0; j < 8; ++j) {
    int s = r0 + j + 8 * hi;
#pragma unroll
    for (int t = 0; t < 4; ++t) {
      int o = n0 + 16 * t + ln;
      if constexpr (EPI == 2)
        ((float*)Cout)[(size_t)s * HID + o] = acc[t][j];
      else if constexpr (EPI == 3)
        ((_Float16*)Cout)[(size_t)o * S_LEN + s] = (_Float16)acc[t][j];
      else
        ((_Float16*)Cout)[(size_t)s * HID + o] = (_Float16)acc[t][j];
    }
  }
}

// ---------------------------------------------------------------------------
// Flash attention (causal), one head x one 64-query block per workgroup.
// K/V 64x64 tiles are TDM-staged into double-buffered LDS; DMA of tile kb+1
// overlaps compute of tile kb (s_wait_tensorcnt + workgroup barriers).
// ---------------------------------------------------------------------------
__global__ void __launch_bounds__(128)
flash_kernel(const _Float16* __restrict__ Qh, const _Float16* __restrict__ Kh,
             const _Float16* __restrict__ Vt, _Float16* __restrict__ Oh) {
  __shared__ _Float16 kbuf[2][64 * 64];   // [key_local][d_local]
  __shared__ _Float16 vbuf[2][64 * 64];   // [d_local][key_local]
  __shared__ _Float16 pt[4][16 * 64];     // per-wave P tile

  const int wave = threadIdx.x >> 5;
  const int lane = threadIdx.x & 31;
  const int ln = lane & 15, hi = lane >> 4;
  const int h  = blockIdx.y;
  const int q0 = blockIdx.x * 64 + wave * 16;
  const int dh = h * HD;
  const int trips = blockIdx.x + 1;       // uniform causal bound (64-key blocks)

  // Q fragments live in registers for the whole kernel.
  v16h qf[2];
  qf[0] = load_frag(Qh, HID, q0, dh + 0,  ln, hi);
  qf[1] = load_frag(Qh, HID, q0, dh + 32, ln, hi);

  v8f oa[4] = {v8f{}, v8f{}, v8f{}, v8f{}};
  float m[8], l[8];
#pragma unroll
  for (int j = 0; j < 8; ++j) { m[j] = -3.0e38f; l[j] = 0.0f; }

  _Float16* pl = &pt[wave][0];
  const float scale = 0.125f;  // 1/sqrt(64)

  if (wave == 0) {  // prime the pipeline with tile 0
    tdm_load_2d(Kh + dh,                  lds_offset_of(&kbuf[0][0]),
                HID, S_LEN, HID, 64, 64);
    tdm_load_2d(Vt + (size_t)dh * S_LEN,  lds_offset_of(&vbuf[0][0]),
                S_LEN, HID, S_LEN, 64, 64);
  }

  for (int kb = 0; kb < trips; ++kb) {
    const int k0 = kb * 64;
    if (wave == 0) __builtin_amdgcn_s_wait_tensorcnt(0);  // tile kb landed
    __syncthreads();                                      // visible to all; prev reads done
    if (wave == 0 && kb + 1 < trips) {                    // overlap: fetch tile kb+1
      const int kn = k0 + 64;
      tdm_load_2d(Kh + (size_t)kn * HID + dh,      lds_offset_of(&kbuf[(kb + 1) & 1][0]),
                  HID, S_LEN, HID, 64, 64);
      tdm_load_2d(Vt + (size_t)dh * S_LEN + kn,    lds_offset_of(&vbuf[(kb + 1) & 1][0]),
                  S_LEN, HID, S_LEN, 64, 64);
    }
    const _Float16* kt = &kbuf[kb & 1][0];
    const _Float16* vt = &vbuf[kb & 1][0];

    // ---- S = Q K^T : 8 WMMAs -> 16 q-rows x 64 keys in sc[0..3]
    v8f sc[4] = {v8f{}, v8f{}, v8f{}, v8f{}};
#pragma unroll
    for (int t = 0; t < 4; ++t) {
#pragma unroll
      for (int c = 0; c < 2; ++c) {
        v16h bf = load_frag(kt, 64, 16 * t, 32 * c, ln, hi);
        sc[t] = wmma_f16(qf[c], bf, sc[t]);
      }
    }

    // ---- scale + causal mask
#pragma unroll
    for (int t = 0; t < 4; ++t)
#pragma unroll
      for (int j = 0; j < 8; ++j) {
        int qr = q0 + j + 8 * hi;
        int ky = k0 + 16 * t + ln;
        float v = sc[t][j] * scale;
        sc[t][j] = (ky > qr) ? -1.0e30f : v;
      }

    // ---- online softmax: row max over 64 keys
    float scl[8];
#pragma unroll
    for (int j = 0; j < 8; ++j) {
      float x = fmaxf(fmaxf(sc[0][j], sc[1][j]), fmaxf(sc[2][j], sc[3][j]));
      x = fmaxf(x, __shfl_xor(x, 1, 32));
      x = fmaxf(x, __shfl_xor(x, 2, 32));
      x = fmaxf(x, __shfl_xor(x, 4, 32));
      x = fmaxf(x, __shfl_xor(x, 8, 32));
      float mn = fmaxf(m[j], x);
      scl[j] = __expf(m[j] - mn);
      m[j] = mn;
    }

    // ---- P = exp(S - m), row sums, rescale running l and O
#pragma unroll
    for (int j = 0; j < 8; ++j) {
      float s = 0.0f;
#pragma unroll
      for (int t = 0; t < 4; ++t) {
        float e = __expf(sc[t][j] - m[j]);
        sc[t][j] = e;
        s += e;
      }
      s += __shfl_xor(s, 1, 32);
      s += __shfl_xor(s, 2, 32);
      s += __shfl_xor(s, 4, 32);
      s += __shfl_xor(s, 8, 32);
      l[j] = l[j] * scl[j] + s;
    }
#pragma unroll
    for (int t = 0; t < 4; ++t)
#pragma unroll
      for (int j = 0; j < 8; ++j) oa[t][j] *= scl[j];

    // ---- reshape P (C-layout) -> A-fragments via wave-private LDS slice
#pragma unroll
    for (int t = 0; t < 4; ++t)
#pragma unroll
      for (int j = 0; j < 8; ++j)
        pl[(j + 8 * hi) * 64 + 16 * t + ln] = (_Float16)sc[t][j];

    v16h pf0 = load_frag((const _Float16*)pl, 64, 0, 0,  ln, hi);
    v16h pf1 = load_frag((const _Float16*)pl, 64, 0, 32, ln, hi);

    // ---- O += P V : 8 WMMAs from the LDS V tile
#pragma unroll
    for (int t = 0; t < 4; ++t) {
      v16h bf0 = load_frag(vt, 64, 16 * t, 0,  ln, hi);
      oa[t] = wmma_f16(pf0, bf0, oa[t]);
      v16h bf1 = load_frag(vt, 64, 16 * t, 32, ln, hi);
      oa[t] = wmma_f16(pf1, bf1, oa[t]);
    }
  }

  // ---- normalize and store O (f16 row-major [s][HID])
#pragma unroll
  for (int j = 0; j < 8; ++j) {
    float inv = 1.0f / l[j];
    int s = q0 + j + 8 * hi;
#pragma unroll
    for (int t = 0; t < 4; ++t)
      Oh[(size_t)s * HID + dh + 16 * t + ln] = (_Float16)(oa[t][j] * inv);
  }
}

// ---------------------------------------------------------------------------
// Host-side orchestration
// ---------------------------------------------------------------------------
extern "C" void kernel_launch(void* const* d_in, const int* in_sizes, int n_in,
                              void* d_out, int out_size, void* d_ws, size_t ws_size,
                              hipStream_t stream) {
  (void)in_sizes; (void)n_in; (void)out_size; (void)ws_size;

  const float* X    = (const float*)d_in[0];
  const float* Wq   = (const float*)d_in[1];
  const float* Wk   = (const float*)d_in[2];
  const float* Wv   = (const float*)d_in[3];
  const float* Wo   = (const float*)d_in[4];
  const float* cosT = (const float*)d_in[5];
  const float* sinT = (const float*)d_in[6];
  // d_in[7] = num_registers (==32, compiled in as NREG)

  char* w = (char*)d_ws;
  const size_t MB = 1024u * 1024u;
  _Float16* Xh  = (_Float16*)(w + 0);              // 4 MB
  _Float16* Wqh = (_Float16*)(w + 4 * MB);         // 0.5 MB each
  _Float16* Wkh = (_Float16*)(w + 4 * MB + 512 * 1024);
  _Float16* Wvh = (_Float16*)(w + 5 * MB);
  _Float16* Woh = (_Float16*)(w + 5 * MB + 512 * 1024);
  _Float16* Qh  = (_Float16*)(w + 6 * MB);         // 4 MB
  _Float16* Kh  = (_Float16*)(w + 10 * MB);        // 4 MB
  _Float16* Vt  = (_Float16*)(w + 14 * MB);        // 4 MB, transposed [o][s]
  _Float16* Oh  = (_Float16*)(w + 18 * MB);        // 4 MB

  const int nX = S_LEN * HID;   // 2,097,152
  const int nW = HID * HID;     // 262,144

  cvt_f32_to_f16<<<(nX / 4 + 255) / 256, 256, 0, stream>>>(X,  Xh,  nX);
  cvt_f32_to_f16<<<(nW / 4 + 255) / 256, 256, 0, stream>>>(Wq, Wqh, nW);
  cvt_f32_to_f16<<<(nW / 4 + 255) / 256, 256, 0, stream>>>(Wk, Wkh, nW);
  cvt_f32_to_f16<<<(nW / 4 + 255) / 256, 256, 0, stream>>>(Wv, Wvh, nW);
  cvt_f32_to_f16<<<(nW / 4 + 255) / 256, 256, 0, stream>>>(Wo, Woh, nW);

  dim3 gg(HID / 64, S_LEN / 64);   // (8, 64)
  gemm_kernel<1><<<gg, 128, 0, stream>>>(Xh, Wqh, Qh, cosT, sinT);       // Q + RoPE
  gemm_kernel<1><<<gg, 128, 0, stream>>>(Xh, Wkh, Kh, cosT, sinT);       // K + RoPE
  gemm_kernel<3><<<gg, 128, 0, stream>>>(Xh, Wvh, Vt, nullptr, nullptr); // V^T

  flash_kernel<<<dim3(S_LEN / 64, NH), 128, 0, stream>>>(Qh, Kh, Vt, Oh);

  gemm_kernel<2><<<gg, 128, 0, stream>>>(Oh, Woh, (float*)d_out, nullptr, nullptr);
}
